// QuantSNN_67224828117633
// MI455X (gfx1250) — 8-bit matrix operand order repaired and checked by reading, compiled, not run on hardware
//
#include <hip/hip_runtime.h>
#include <stdint.h>

typedef int v8i __attribute__((ext_vector_type(8)));

#define T_STEPS 50
#define BATCH   256
#define IN_DIM  2048
#define H_DIM   4096
#define OUT_DIM 512

__global__ void snn_init_kernel(float* p, long n) {
    long stride = (long)gridDim.x * blockDim.x;
    for (long i = (long)blockIdx.x * blockDim.x + threadIdx.x; i < n; i += stride)
        p[i] = 0.0f;
}

__global__ void snn_absmax_kernel(const float* __restrict__ w, long n, unsigned* out) {
    __shared__ float red[256];
    long stride = (long)gridDim.x * blockDim.x;
    float m = 0.0f;
    for (long i = (long)blockIdx.x * blockDim.x + threadIdx.x; i < n; i += stride)
        m = fmaxf(m, fabsf(w[i]));
    red[threadIdx.x] = m;
    __syncthreads();
    for (int s = 128; s > 0; s >>= 1) {
        if ((int)threadIdx.x < s)
            red[threadIdx.x] = fmaxf(red[threadIdx.x], red[threadIdx.x + s]);
        __syncthreads();
    }
    if (threadIdx.x == 0)
        atomicMax(out, __float_as_uint(red[0]));
}

__global__ void snn_quant_kernel(const float* __restrict__ w,
                                 const unsigned* __restrict__ absmax,
                                 int8_t* __restrict__ w8, long n) {
    float amax  = __uint_as_float(*absmax);
    float scale = amax / 127.0f;
    float inv   = (scale > 0.0f) ? (1.0f / scale) : 0.0f;
    long stride = (long)gridDim.x * blockDim.x;
    for (long i = (long)blockIdx.x * blockDim.x + threadIdx.x; i < n; i += stride) {
        float q = rintf(w[i] * inv);
        q = fminf(fmaxf(q, -128.0f), 127.0f);
        w8[i] = (int8_t)q;
    }
}

__global__ void snn_f2i8_kernel(const float* __restrict__ s, int8_t* __restrict__ s8, long n) {
    long stride = (long)gridDim.x * blockDim.x;
    for (long i = (long)blockIdx.x * blockDim.x + threadIdx.x; i < n; i += stride)
        s8[i] = (s[i] != 0.0f) ? (int8_t)1 : (int8_t)0;
}

__device__ __forceinline__ v8i snn_loadB(const int8_t* p) {
    int4 lo = *(const int4*)(p);
    int4 hi = *(const int4*)(p + 32);
    v8i r;
    r[0] = lo.x; r[1] = lo.y; r[2] = lo.z; r[3] = lo.w;
    r[4] = hi.x; r[5] = hi.y; r[6] = hi.z; r[7] = hi.w;
    return r;
}
__device__ __forceinline__ v8i snn_loadA(const int8_t* p) {
    int2 a0 = *(const int2*)(p);
    int2 a1 = *(const int2*)(p + 16);
    int2 a2 = *(const int2*)(p + 32);
    int2 a3 = *(const int2*)(p + 48);
    v8i r;
    r[0] = a0.x; r[1] = a0.y; r[2] = a1.x; r[3] = a1.y;
    r[4] = a2.x; r[5] = a2.y; r[6] = a3.x; r[7] = a3.y;
    return r;
}

__global__ __launch_bounds__(256)
void snn_gemm_lif_kernel(const int8_t* __restrict__ A8,
                         const int8_t* __restrict__ W8,
                         const unsigned* __restrict__ absmax,
                         const float* __restrict__ bias,
                         const float* __restrict__ tau,
                         float* __restrict__ v,
                         int8_t* __restrict__ sp8,
                         float*  __restrict__ spf,
                         int N, int K) {
    const int lane = threadIdx.x & 31;
    const int wave = threadIdx.x >> 5;
    const int l15  = lane & 15;
    const int half = lane >> 4;

    const int n  = blockIdx.x * 128 + wave * 16 + l15;
    const int m0 = blockIdx.y * 64;

    const int8_t* bptr = W8 + (long)n * K + 8 * half;
    const int8_t* aptr = A8 + (long)(m0 + l15) * K + 8 * half;
    const long arow = (long)16 * K;

    v8i acc[4];
#pragma unroll
    for (int j = 0; j < 4; ++j)
#pragma unroll
        for (int e = 0; e < 8; ++e) acc[j][e] = 0;

    v8i Bc = snn_loadA(bptr);
    v8i Ac[4];
#pragma unroll
    for (int j = 0; j < 4; ++j) Ac[j] = snn_loadA(aptr + j * arow);

    for (int k = 64; k < K; k += 64) {
        v8i Bn = snn_loadA(bptr + k);
        v8i An[4];
#pragma unroll
        for (int j = 0; j < 4; ++j) An[j] = snn_loadA(aptr + j * arow + k);
        __builtin_prefetch(bptr + k + 64, 0, 1);

#pragma unroll
        for (int j = 0; j < 4; ++j)
            acc[j] = __builtin_amdgcn_wmma_i32_16x16x64_iu8(
                false, Ac[j], true, Bc, acc[j], false, false);

        Bc = Bn;
#pragma unroll
        for (int j = 0; j < 4; ++j) Ac[j] = An[j];
    }

#pragma unroll
    for (int j = 0; j < 4; ++j)
        acc[j] = __builtin_amdgcn_wmma_i32_16x16x64_iu8(
            false, Ac[j], true, Bc, acc[j], false, false);

    const float scale = __uint_as_float(*absmax) / 127.0f;
    const float bn    = bias[n];
    const float itau  = 1.0f / fmaxf(tau[n], 1.0f);

#pragma unroll
    for (int j = 0; j < 4; ++j) {
#pragma unroll
        for (int r = 0; r < 8; ++r) {
            int m = m0 + 16 * j + r + 8 * half;
            long idx = (long)m * N + n;
            float cur = scale * (float)acc[j][r] + bn;
            float vo  = v[idx];
            float vn_ = vo + itau * (cur - vo);
            bool  sp  = vn_ > 1.0f;
            v[idx] = sp ? 0.0f : vn_;
            if (sp8) sp8[idx] = sp ? (int8_t)1 : (int8_t)0;
            if (spf) spf[idx] = sp ? 1.0f : 0.0f;
        }
    }
}

extern "C" void kernel_launch(void* const* d_in, const int* in_sizes, int n_in,
                              void* d_out, int out_size, void* d_ws, size_t ws_size,
                              hipStream_t stream) {
    const float* s    = (const float*)d_in[0];
    const float* W1   = (const float*)d_in[1];
    const float* b1   = (const float*)d_in[2];
    const float* tau1 = (const float*)d_in[3];
    const float* W2   = (const float*)d_in[4];
    const float* b2   = (const float*)d_in[5];
    const float* tau2 = (const float*)d_in[6];
    const float* W3   = (const float*)d_in[7];
    const float* b3   = (const float*)d_in[8];
    const float* tau3 = (const float*)d_in[9];
    float* out = (float*)d_out;

    char* ws = (char*)d_ws;
    unsigned* absmax = (unsigned*)ws;
    float* v1 = (float*)(ws + 256);
    float* v2 = v1 + (size_t)BATCH * H_DIM;
    float* v3 = v2 + (size_t)BATCH * H_DIM;
    int8_t* s1_8 = (int8_t*)(v3 + (size_t)BATCH * OUT_DIM);
    int8_t* s2_8 = s1_8 + (size_t)BATCH * H_DIM;
    int8_t* w1_8 = s2_8 + (size_t)BATCH * H_DIM;
    int8_t* w2_8 = w1_8 + (size_t)H_DIM * IN_DIM;
    int8_t* w3_8 = w2_8 + (size_t)H_DIM * H_DIM;
    int8_t* sq8  = w3_8 + (size_t)OUT_DIM * H_DIM;

    long nzero = 64 + (long)BATCH * H_DIM * 2 + (long)BATCH * OUT_DIM;
    snn_init_kernel<<<2048, 256, 0, stream>>>((float*)ws, nzero);

    snn_absmax_kernel<<<1024, 256, 0, stream>>>(W1, (long)H_DIM * IN_DIM,  absmax + 0);
    snn_absmax_kernel<<<1024, 256, 0, stream>>>(W2, (long)H_DIM * H_DIM,   absmax + 1);
    snn_absmax_kernel<<<1024, 256, 0, stream>>>(W3, (long)OUT_DIM * H_DIM, absmax + 2);

    snn_quant_kernel<<<2048, 256, 0, stream>>>(W1, absmax + 0, w1_8, (long)H_DIM * IN_DIM);
    snn_quant_kernel<<<2048, 256, 0, stream>>>(W2, absmax + 1, w2_8, (long)H_DIM * H_DIM);
    snn_quant_kernel<<<2048, 256, 0, stream>>>(W3, absmax + 2, w3_8, (long)OUT_DIM * H_DIM);

    snn_f2i8_kernel<<<4096, 256, 0, stream>>>(s, sq8, (long)T_STEPS * BATCH * IN_DIM);

    dim3 gridH(H_DIM / 128, BATCH / 64);
    dim3 gridO(OUT_DIM / 128, BATCH / 64);

    for (int t = 0; t < T_STEPS; ++t) {
        const int8_t* st = sq8 + (size_t)t * BATCH * IN_DIM;
        snn_gemm_lif_kernel<<<gridH, 256, 0, stream>>>(
            st,   w1_8, absmax + 0, b1, tau1, v1, s1_8, nullptr, H_DIM, IN_DIM);
        snn_gemm_lif_kernel<<<gridH, 256, 0, stream>>>(
            s1_8, w2_8, absmax + 1, b2, tau2, v2, s2_8, nullptr, H_DIM, H_DIM);
        snn_gemm_lif_kernel<<<gridO, 256, 0, stream>>>(
            s2_8, w3_8, absmax + 2, b3, tau3, v3, nullptr,
            out + (size_t)t * BATCH * OUT_DIM, OUT_DIM, H_DIM);
    }
}
